// CAttentionBlock_72756745994848
// MI455X (gfx1250) — matrix operand order repaired and checked by reading, not compiled, not run on hardware
//
#include <hip/hip_runtime.h>
#include <math.h>

// ---------------------------------------------------------------------------
// Types for CDNA5 WMMA (wave32, bf16 16x16x32, f32 accumulate)
// ---------------------------------------------------------------------------
typedef unsigned short u16;
typedef __bf16 bf16_t;
typedef bf16_t v16bf __attribute__((ext_vector_type(16)));
typedef u16    v8u   __attribute__((ext_vector_type(8)));
typedef float  v8f   __attribute__((ext_vector_type(8)));

#define BB    4
#define HH    128
#define WW    128
#define CC    256
#define NHEAD 8
#define DH    32
#define HIDN  1024
#define NTOK  65536      // tokens per branch (B*H*W)
#define MTILE 128        // token tile per workgroup (= 32 windows)
#define HCH   64         // hidden chunk per fused-MLP iteration
#define SA    264        // LDS row stride (elements) for K=256 tiles (pad 8)
#define SH    72         // LDS row stride (elements) for K=64 tiles  (pad 8)

__device__ __forceinline__ u16 f2bf(float f) {
  union { float f; unsigned u; } x; x.f = f;
  unsigned r = x.u + 0x7fffu + ((x.u >> 16) & 1u);   // round-to-nearest-even
  return (u16)(r >> 16);
}

__device__ __forceinline__ float bf2f(u16 h) {
  union { unsigned u; float f; } x; x.u = ((unsigned)h) << 16;
  return x.f;
}

union Frag { v16bf v; v8u h[2]; };

// A fragment (16x32 bf16, row-major source): lane L holds row (L&15);
// lanes 0-15 -> K {0..7,16..23}, lanes 16-31 -> K {8..15,24..31}.
__device__ __forceinline__ v16bf load_afrag(const u16* base, int stride, int k0, int lane) {
  int row = lane & 15;
  int kb  = (lane >> 4) << 3;
  const u16* p = base + row * stride + k0 + kb;
  Frag u;
  u.h[0] = *(const v8u*)p;
  u.h[1] = *(const v8u*)(p + 16);
  return u.v;
}

// B fragment (32x16 bf16) from an N-major [n][k] LDS buffer: lane L holds
// column (L&15); lanes 0-15 -> K {0..7,16..23}, lanes 16-31 -> K {8..15,24..31}.
__device__ __forceinline__ v16bf load_bfrag(const u16* base, int stride, int k0, int lane) {
  int n  = lane & 15;
  int kb = (lane >> 4) << 3;
  const u16* p = base + n * stride + k0 + kb;
  Frag u;
  u.h[0] = *(const v8u*)p;
  u.h[1] = *(const v8u*)(p + 16);
  return u.v;
}

__device__ __forceinline__ v8f wmma_bf16(v16bf a, v16bf b, v8f c) {
  return __builtin_amdgcn_wmma_f32_16x16x32_bf16(false, a, false, b, (short)0, c, false, false);
}

__device__ __forceinline__ float gelu_exact(float x) {
  return 0.5f * x * (1.0f + erff(x * 0.70710678118654752f));
}

// ---------------------------------------------------------------------------
// Kernel 1: convert+transpose weights to bf16.
//   w1 [C][HID] f32  -> w1t [HID][C] bf16   (B of GEMM1, n-major)
//   w2 [HID][C] f32  -> w2t [C][HID] bf16   (B of GEMM2, n-major)
// ---------------------------------------------------------------------------
__global__ __launch_bounds__(256) void prep_weights_kernel(
    const float* w1_0, const float* w1_1, const float* w1_2, const float* w1_3,
    const float* w2_0, const float* w2_1, const float* w2_2, const float* w2_3,
    u16* w1t, u16* w2t)
{
  const int per = CC * HIDN;                    // 262144 = 2^18
  int id = blockIdx.x * 256 + threadIdx.x;      // 0 .. 8*per-1
  if (id < 4 * per) {                           // w1 path
    int br = id >> 18;
    int e  = id & (per - 1);
    int n  = e >> 8;                            // hidden index
    int k  = e & 255;                           // C index
    const float* w1 = br == 0 ? w1_0 : br == 1 ? w1_1 : br == 2 ? w1_2 : w1_3;
    w1t[(size_t)br * per + (size_t)n * CC + k] = f2bf(w1[(size_t)k * HIDN + n]);
  } else {                                      // w2 path
    int m  = id - 4 * per;
    int br = m >> 18;
    int e  = m & (per - 1);
    int n  = e >> 10;                           // C index
    int k  = e & 1023;                          // hidden index
    const float* w2 = br == 0 ? w2_0 : br == 1 ? w2_1 : br == 2 ? w2_2 : w2_3;
    w2t[(size_t)br * per + (size_t)n * HIDN + k] = f2bf(w2[(size_t)k * CC + n]);
  }
}

// ---------------------------------------------------------------------------
// Kernel 2 (fused): windowed cross-attention (ws=2) + MLP + residual,
// scattered back through window-unpartition into [B,H,W,C].
//
// Workgroup = (128-token tile == 32 windows) x (branch), 8 waves.
//   Phase A: each wave computes attention for 4 windows x 8 heads
//            (lane = channel-in-head, exact f32 softmax via shuffles)
//            and writes the result straight into the LDS A-tile as bf16.
//   Phase B: fused GEMM1(K=256) -> GELU -> GEMM2(K=1024 in 16 chunks of 64)
//            with bf16 WMMA, f32 accumulators held in registers.
//   Epilogue: + b2 + residual (from LDS A-tile), window-unpartition scatter.
// No intermediate activation ever touches HBM.
// ---------------------------------------------------------------------------
__global__ __launch_bounds__(256) void fused_cattn_mlp_kernel(
    const float* __restrict__ r, const float* __restrict__ g,
    const float* __restrict__ b, const float* __restrict__ ir,
    const u16* __restrict__ w1t_all, const u16* __restrict__ w2t_all,
    const float* b1_0, const float* b1_1, const float* b1_2, const float* b1_3,
    const float* b2_0, const float* b2_1, const float* b2_2, const float* b2_3,
    float* __restrict__ out)
{
  extern __shared__ __align__(16) char smem_raw[];
  u16* As  = (u16*)smem_raw;            // [128][SA]  attention out (tokens x C), bf16
  u16* W1c = As  + MTILE * SA;          // [64][SA]   W1 chunk, n-major [n][k=C]
  u16* W2c = W1c + HCH * SA;            // [256][SH]  W2 chunk, n-major [n][k=HCH]
  u16* Hs  = W2c + CC * SH;             // [128][SH]  GELU'd hidden chunk

  int br   = blockIdx.y;
  int tile = blockIdx.x;                // 0..511 (128 tokens each)
  const u16* W1t = w1t_all + (size_t)br * HIDN * CC;   // [HID][C]
  const u16* W2t = w2t_all + (size_t)br * CC * HIDN;   // [C][HID]
  const float* b1p = br == 0 ? b1_0 : br == 1 ? b1_1 : br == 2 ? b1_2 : b1_3;
  const float* b2p = br == 0 ? b2_0 : br == 1 ? b2_1 : br == 2 ? b2_2 : b2_3;
  // branch q/kv pairing: 0:(r,g) 1:(g,b) 2:(b,ir) 3:(ir,g)
  const float* Q = br == 0 ? r : br == 1 ? g : br == 2 ? b : ir;
  const float* K = br == 0 ? g : br == 1 ? b : br == 2 ? ir : g;

  int tid  = threadIdx.x;
  int lane = tid & 31;
  int wv   = tid >> 5;

  // ---------------- Phase A: attention into LDS (bf16) --------------------
  // Wave wv owns windows [tile*32 + wv*4, +4) -> token rows [wv*16, wv*16+16).
  const float scale = 0.17677669529663687f;   // 1/sqrt(32)
  int winbase = tile * 32;
  for (int pair = 0; pair < 32; pair++) {     // 4 windows * 8 heads
    int wl   = wv * 4 + (pair >> 3);          // local window (0..31)
    int head = pair & 7;
    int win  = winbase + wl;                  // window within branch
    int bb = win >> 12;                       // 64*64 windows per batch image
    int wy = (win >> 6) & 63;
    int wx = win & 63;
    int cch = head * DH + lane;

    float q[4], kv[4];
#pragma unroll
    for (int i = 0; i < 4; i++) {
      int h = wy * 2 + (i >> 1);
      int w = wx * 2 + (i & 1);
      size_t idx = (((size_t)bb * HH + h) * WW + w) * CC + cch;
      q[i]  = Q[idx];
      kv[i] = K[idx];
    }

    float s[4][4];
#pragma unroll
    for (int i = 0; i < 4; i++)
#pragma unroll
      for (int j = 0; j < 4; j++)
        s[i][j] = q[i] * kv[j];
#pragma unroll
    for (int off = 16; off >= 1; off >>= 1) {
#pragma unroll
      for (int i = 0; i < 4; i++)
#pragma unroll
        for (int j = 0; j < 4; j++)
          s[i][j] += __shfl_xor(s[i][j], off, 32);
    }

#pragma unroll
    for (int i = 0; i < 4; i++) {
      float m = s[i][0] * scale;
#pragma unroll
      for (int j = 1; j < 4; j++) m = fmaxf(m, s[i][j] * scale);
      float p[4], sum = 0.f;
#pragma unroll
      for (int j = 0; j < 4; j++) { p[j] = expf(s[i][j] * scale - m); sum += p[j]; }
      float inv = 1.0f / sum;
      float o = 0.f;
#pragma unroll
      for (int j = 0; j < 4; j++) o += p[j] * inv * kv[j];   // v == kv
      As[(wl * 4 + i) * SA + cch] = f2bf(o);
    }
  }
  __syncthreads();

  // ---------------- Phase B: fused MLP (bf16 WMMA, f32 acc) ----------------
  // Persistent output accumulators: wave owns a 64x64 sub-tile of [128x256]
  int mblk = (wv >> 2) * 64;
  int nblk = (wv & 3) * 64;
  v8f zacc[4][4];
#pragma unroll
  for (int mt = 0; mt < 4; mt++)
#pragma unroll
    for (int nt = 0; nt < 4; nt++)
#pragma unroll
      for (int rr = 0; rr < 8; rr++) zacc[mt][nt][rr] = 0.0f;

  for (int hc = 0; hc < HIDN; hc += HCH) {
    // Load W1 chunk rows [hc, hc+64) and W2 chunk cols [hc, hc+64)
    for (int e = tid; e < HCH * CC; e += 256) {
      int n = e >> 8, k = e & 255;
      W1c[n * SA + k] = W1t[(size_t)(hc + n) * CC + k];
    }
    for (int e = tid; e < CC * HCH; e += 256) {
      int n = e >> 6, k = e & 63;
      W2c[n * SH + k] = W2t[(size_t)n * HIDN + hc + k];
    }
    __syncthreads();

    // ---- Stage 1: H = GELU(A @ W1chunk + b1), wave handles 16 rows x 64 cols
    int m16 = wv * 16;
    v8f hacc[4];
#pragma unroll
    for (int nt = 0; nt < 4; nt++)
#pragma unroll
      for (int rr = 0; rr < 8; rr++) hacc[nt][rr] = 0.0f;

    for (int k0 = 0; k0 < CC; k0 += 32) {
      v16bf a = load_afrag(As + m16 * SA, SA, k0, lane);
#pragma unroll
      for (int nt = 0; nt < 4; nt++) {
        v16bf bf = load_bfrag(W1c + (nt * 16) * SA, SA, k0, lane);
        hacc[nt] = wmma_bf16(a, bf, hacc[nt]);
      }
    }
#pragma unroll
    for (int nt = 0; nt < 4; nt++) {
      int colg = hc + nt * 16 + (lane & 15);
      float bias = b1p[colg];
#pragma unroll
      for (int rr = 0; rr < 8; rr++) {
        float x = hacc[nt][rr] + bias;
        int row = m16 + rr + ((lane >> 4) << 3);
        Hs[row * SH + nt * 16 + (lane & 15)] = f2bf(gelu_exact(x));
      }
    }
    __syncthreads();

    // ---- Stage 2: Z += Hchunk @ W2chunk (K = 64)
    for (int k0 = 0; k0 < HCH; k0 += 32) {
      v16bf a0[4];
#pragma unroll
      for (int mt = 0; mt < 4; mt++)
        a0[mt] = load_afrag(Hs + (mblk + mt * 16) * SH, SH, k0, lane);
#pragma unroll
      for (int nt = 0; nt < 4; nt++) {
        v16bf bf = load_bfrag(W2c + (nblk + nt * 16) * SH, SH, k0, lane);
#pragma unroll
        for (int mt = 0; mt < 4; mt++)
          zacc[mt][nt] = wmma_bf16(a0[mt], bf, zacc[mt][nt]);
      }
    }
    __syncthreads();   // before next chunk overwrites W1c/W2c/Hs
  }

  // -------- Epilogue: + b2 + residual (LDS), window-unpartition scatter ----
#pragma unroll
  for (int mt = 0; mt < 4; mt++) {
#pragma unroll
    for (int nt = 0; nt < 4; nt++) {
      int col = nblk + nt * 16 + (lane & 15);
      float bias = b2p[col];
#pragma unroll
      for (int rr = 0; rr < 8; rr++) {
        int lm = mblk + mt * 16 + rr + ((lane >> 4) << 3);
        int t  = tile * MTILE + lm;                 // token in window order
        float val = zacc[mt][nt][rr] + bias + bf2f(As[lm * SA + col]);
        int win = t >> 2, i = t & 3;
        int bb  = win >> 12;
        int wy  = (win >> 6) & 63;
        int wx  = win & 63;
        int h   = wy * 2 + (i >> 1);
        int w   = wx * 2 + (i & 1);
        out[(size_t)br * (BB * HH * WW * CC)
            + (((size_t)bb * HH + h) * WW + w) * CC + col] = val;
      }
    }
  }
}

// ---------------------------------------------------------------------------
// Launch.  Workspace: only bf16 transposed weights (4 MB total).
//   [0, 2MB)    w1t bf16 (4 branches, [HID][C])
//   [2MB, 4MB)  w2t bf16 (4 branches, [C][HID])
// ---------------------------------------------------------------------------
extern "C" void kernel_launch(void* const* d_in, const int* in_sizes, int n_in,
                              void* d_out, int out_size, void* d_ws, size_t ws_size,
                              hipStream_t stream) {
  (void)in_sizes; (void)n_in; (void)out_size; (void)ws_size;
  const float* r  = (const float*)d_in[0];
  const float* g  = (const float*)d_in[1];
  const float* b  = (const float*)d_in[2];
  const float* ir = (const float*)d_in[3];
  // branch weights: d_in[4 + 4*br + {0:w1,1:b1,2:w2,3:b2}]
  const float* w1[4]; const float* b1[4]; const float* w2[4]; const float* b2[4];
  for (int br = 0; br < 4; br++) {
    w1[br] = (const float*)d_in[4 + 4 * br + 0];
    b1[br] = (const float*)d_in[4 + 4 * br + 1];
    w2[br] = (const float*)d_in[4 + 4 * br + 2];
    b2[br] = (const float*)d_in[4 + 4 * br + 3];
  }

  char* ws = (char*)d_ws;
  const size_t wmat_bytes = (size_t)4 * CC * HIDN * sizeof(u16);   // 2097152
  u16* w1t = (u16*)ws;
  u16* w2t = (u16*)(ws + wmat_bytes);

  // 1) weights -> bf16 transposed
  prep_weights_kernel<<<8192, 256, 0, stream>>>(
      w1[0], w1[1], w1[2], w1[3], w2[0], w2[1], w2[2], w2[3], w1t, w2t);

  // 2) fused attention + MLP: 512 token-tiles x 4 branches, 153KB dynamic LDS
  const size_t lds_bytes =
      (size_t)(MTILE * SA + HCH * SA + CC * SH + MTILE * SH) * sizeof(u16);
  fused_cattn_mlp_kernel<<<dim3(512, 4), 256, lds_bytes, stream>>>(
      r, g, b, ir, w1t, w2t,
      b1[0], b1[1], b1[2], b1[3],
      b2[0], b2[1], b2[2], b2[3],
      (float*)d_out);
}
